// AttentionEinOps_33346126086635
// MI455X (gfx1250) — matrix operand order repaired and checked by reading, not compiled, not run on hardware
//
#include <hip/hip_runtime.h>
#include <cmath>

// ---------------------------------------------------------------------------
// MI455X / gfx1250 (CDNA5, wave32) fused attention with per-head LayerNorm.
// All GEMMs use v_wmma_f32_16x16x32_bf16 (bf16 in, f32 accumulate).
// k_attn stages K/V chunks into LDS with global_load_async_to_lds_b128
// (ASYNCcnt), double-buffered so the DMA overlaps the WMMA work.
// Problem: B=2, SQ=SK=2048, D=1024, N=16, H=64.
// ---------------------------------------------------------------------------

typedef __bf16 bf16;
typedef __attribute__((ext_vector_type(16))) bf16  v16bf;
typedef __attribute__((ext_vector_type(8)))  bf16  v8bf;
typedef __attribute__((ext_vector_type(4)))  bf16  v4bf;
typedef __attribute__((ext_vector_type(8)))  float v8f;

#define BB 2
#define SQL 2048
#define SKL 2048
#define DD 1024
#define NH 16
#define HH 64

// LDS row paddings: odd multiples of 16B keep the 16-lane b128 fragment
// reads on distinct bank groups.
#define KP 72   // K tile row: 64 bf16 + 8 pad  (144B stride)
#define VP 40   // V tile row: 32 bf16 + 8 pad  (80B stride)
#define PP 40   // P tile row: 32 bf16 + 8 pad  (80B stride)

__device__ inline v8f v8f_zero() {
    v8f z;
#pragma unroll
    for (int i = 0; i < 8; ++i) z[i] = 0.0f;
    return z;
}

__device__ inline v8f wmma_bf16(v16bf a, v16bf b, v8f c) {
    // (neg_a, A, neg_b, B, c_mod, C, reuse_a, reuse_b)
    return __builtin_amdgcn_wmma_f32_16x16x32_bf16(false, a, false, b,
                                                   (short)0, c, false, false);
}

__device__ inline v16bf cat8(v8bf lo, v8bf hi) {
    return __builtin_shufflevector(lo, hi, 0, 1, 2, 3, 4, 5, 6, 7,
                                   8, 9, 10, 11, 12, 13, 14, 15);
}

// A-fragment (16x32 bf16): lane L -> row = base + (L&15); the 16 bf16 per lane
// live at K = half*16 + [0..15] (contiguous).
__device__ inline v16bf load_afrag(const bf16* __restrict__ row, int half) {
    v8bf lo = *(const v8bf*)(row + half * 16);
    v8bf hi = *(const v8bf*)(row + half * 16 + 8);
    return cat8(lo, hi);
}

// B-fragment from LDS: 16 consecutive bf16 as two 16B loads (16B-aligned).
__device__ inline v16bf ld_lds_frag(const bf16* p) {
    return cat8(*(const v8bf*)p, *(const v8bf*)(p + 8));
}

// Async DMA: copy 16B global -> LDS, tracked by ASYNCcnt (ISA §15.18 op 98).
// Low 32 bits of a flat pointer into LDS are the LDS byte address (§10.2).
__device__ inline void async_ld16(const bf16* lds_dst, const bf16* gsrc) {
    unsigned l = (unsigned)(uintptr_t)lds_dst;
    asm volatile("global_load_async_to_lds_b128 %0, %1, off"
                 :: "v"(l), "v"(gsrc) : "memory");
}

// ---------------------------------------------------------------------------
// Stage 0: pack fp32 -> bf16 (activations + transposed weights)
// ---------------------------------------------------------------------------
__global__ __launch_bounds__(256) void k_cvt_bf16(const float* __restrict__ in,
                                                  bf16* __restrict__ out, int n) {
    int i = (blockIdx.x * 256 + threadIdx.x) * 4;
    if (i + 3 < n) {
        float4 v = *(const float4*)(in + i);
        v4bf o;
        o[0] = (bf16)v.x; o[1] = (bf16)v.y; o[2] = (bf16)v.z; o[3] = (bf16)v.w;
        *(v4bf*)(out + i) = o;
    }
}

// W_{Q,K,V}[N,D,H] -> Wb[proj][n*H+h][D]  (col-major B for the projection GEMM)
__global__ __launch_bounds__(256) void k_pack_w(const float* __restrict__ Wq,
                                                const float* __restrict__ Wk,
                                                const float* __restrict__ Wv,
                                                bf16* __restrict__ Wb) {
    const float* W = (blockIdx.y == 0) ? Wq : (blockIdx.y == 1) ? Wk : Wv;
    bf16* dst = Wb + (size_t)blockIdx.y * DD * (NH * HH);
    int o = blockIdx.x * 256 + threadIdx.x;       // [0, 1048576)
    int c = o >> 10;                              // n*H + h
    int d = o & 1023;
    int n = c >> 6, h = c & 63;
    dst[o] = (bf16)W[((size_t)n * DD + d) * HH + h];
}

// W_O[N,H,D] -> WOb[d][n*H+h]  (col-major B for the output GEMM)
__global__ __launch_bounds__(256) void k_pack_wo(const float* __restrict__ WO,
                                                 bf16* __restrict__ WOb) {
    int o = blockIdx.x * 256 + threadIdx.x;
    int d = o >> 10, k = o & 1023;
    WOb[o] = (bf16)WO[(size_t)k * DD + d];
}

// ---------------------------------------------------------------------------
// Stage 1: QKV projection + bias + per-head LayerNorm (Q,K).
// grid = (256 row-tiles, 3 projections), block = 256 (8 waves).
// Wave w owns heads {2w, 2w+1}: 8 output tiles of 16x16 over 16 rows.
// ---------------------------------------------------------------------------
__global__ __launch_bounds__(256) void k_proj(
    const bf16* __restrict__ xqb, const bf16* __restrict__ xkvb,
    const bf16* __restrict__ Wb,
    const float* __restrict__ bQ, const float* __restrict__ bK,
    const float* __restrict__ bV,
    const float* __restrict__ ln1g, const float* __restrict__ ln1b,
    const float* __restrict__ ln2g, const float* __restrict__ ln2b,
    float* __restrict__ out_k, float* __restrict__ out_v,
    bf16* __restrict__ Qb, bf16* __restrict__ Kb, bf16* __restrict__ Vt) {
    const int proj = blockIdx.y;                  // 0=Q 1=K 2=V
    const int rowTile = blockIdx.x;               // 16 rows each, 4096 rows total
    const int lane = threadIdx.x & 31;
    const int wave = threadIdx.x >> 5;
    const int half = lane >> 4;
    const int l16 = lane & 15;

    const bf16* X = (proj == 0) ? xqb : xkvb;
    const bf16* W = Wb + (size_t)proj * DD * (NH * HH);
    const float* biasP = (proj == 0) ? bQ : (proj == 1) ? bK : bV;
    const float* lng = (proj == 0) ? ln1g : ln2g;
    const float* lnb = (proj == 0) ? ln1b : ln2b;

    const int row0 = rowTile * 16;
    const int b = row0 >> 11;                     // batch (SQ = 2048)
    const int srow = row0 & 2047;
    const bf16* arow = X + (size_t)(row0 + l16) * DD;

    v8f acc[8];
#pragma unroll
    for (int t = 0; t < 8; ++t) acc[t] = v8f_zero();

    for (int k0 = 0; k0 < DD; k0 += 32) {
        __builtin_prefetch(arow + k0 + 128, 0, 0);   // global_prefetch_b8
        v16bf a = load_afrag(arow + k0, half);
#pragma unroll
        for (int t = 0; t < 8; ++t) {
            int n = 2 * wave + (t >> 2);
            int c = n * HH + (t & 3) * 16 + l16;     // output column
            v16bf bf = *(const v16bf*)(W + (size_t)c * DD + k0 + half * 16);
            acc[t] = wmma_bf16(a, bf, acc[t]);
        }
    }

#pragma unroll
    for (int hd = 0; hd < 2; ++hd) {
        const int n = 2 * wave + hd;
        float bias[4], g[4], be[4];
#pragma unroll
        for (int ht = 0; ht < 4; ++ht) {
            int h = ht * 16 + l16;
            bias[ht] = biasP[n * HH + h];
            g[ht] = (proj < 2) ? lng[h] : 0.f;
            be[ht] = (proj < 2) ? lnb[h] : 0.f;
        }
#pragma unroll
        for (int ht = 0; ht < 4; ++ht)
#pragma unroll
            for (int r = 0; r < 8; ++r) acc[hd * 4 + ht][r] += bias[ht];

        if (proj < 2) {
            // LayerNorm over H=64: lane-local partials over the 4 h-tiles,
            // then a 16-lane half-wave butterfly (rows live per half).
#pragma unroll
            for (int r = 0; r < 8; ++r) {
                float x0 = acc[hd * 4 + 0][r], x1 = acc[hd * 4 + 1][r];
                float x2 = acc[hd * 4 + 2][r], x3 = acc[hd * 4 + 3][r];
                float s1 = x0 + x1 + x2 + x3;
                float s2 = x0 * x0 + x1 * x1 + x2 * x2 + x3 * x3;
#pragma unroll
                for (int m = 1; m < 16; m <<= 1) {
                    s1 += __shfl_xor(s1, m, 32);
                    s2 += __shfl_xor(s2, m, 32);
                }
                float mean = s1 * (1.0f / 64.0f);
                float var = s2 * (1.0f / 64.0f) - mean * mean;
                float inv = rsqrtf(var + 1e-5f);
#pragma unroll
                for (int ht = 0; ht < 4; ++ht) {
                    int t = hd * 4 + ht;
                    acc[t][r] = (acc[t][r] - mean) * inv * g[ht] + be[ht];
                }
            }
        }

#pragma unroll
        for (int ht = 0; ht < 4; ++ht) {
            int t = hd * 4 + ht;
            int h = ht * 16 + l16;
#pragma unroll
            for (int r = 0; r < 8; ++r) {
                int m = r + 8 * half;
                int s = srow + m;
                float v = acc[t][r];
                if (proj == 0) {
                    Qb[(((size_t)b * NH + n) * SQL + s) * HH + h] = (bf16)v;
                } else if (proj == 1) {
                    out_k[(((size_t)b * SKL + s) * NH + n) * HH + h] = v;
                    Kb[(((size_t)b * NH + n) * SKL + s) * HH + h] = (bf16)v;
                } else {
                    out_v[(((size_t)b * SKL + s) * NH + n) * HH + h] = v;
                    Vt[(((size_t)b * NH + n) * HH + h) * SKL + s] = (bf16)v;
                }
            }
        }
    }
}

// ---------------------------------------------------------------------------
// Stage 2: causal flash attention.
// One block = 8 waves = 128 consecutive q-rows of ONE head, so K/V chunks are
// shared block-wide: each 32-key chunk of K (32x64) and V (64x32) is staged
// into LDS by async DMA (global_load_async_to_lds_b128, ASYNCcnt),
// double-buffered to overlap with the 8 WMMAs per chunk.
// P goes through per-wave LDS to move from C-layout to A-layout (bf16).
// ---------------------------------------------------------------------------
__global__ __launch_bounds__(256) void k_attn(const bf16* __restrict__ Qb,
                                              const bf16* __restrict__ Kb,
                                              const bf16* __restrict__ Vt,
                                              bf16* __restrict__ zb) {
    __shared__ __align__(64) bf16 kbuf[2][32][KP];   // [s_local][h]
    __shared__ __align__(64) bf16 vbuf[2][64][VP];   // [h][s_local]
    __shared__ __align__(64) bf16 plds[8][16][PP];   // per-wave P tile

    const int tid = threadIdx.x;
    const int lane = tid & 31;
    const int wave = tid >> 5;
    const int half = lane >> 4;
    const int l16 = lane & 15;

    const int blk = blockIdx.x;                   // 512 = B * N * (SQ/128)
    const int b = blk >> 8;
    const int n = (blk >> 4) & 15;
    const int qb = blk & 15;
    const int q0 = qb * 128 + wave * 16;          // this wave's q-tile

    const bf16* Khead = Kb + ((size_t)b * NH + n) * SKL * HH;
    const bf16* Vhead = Vt + ((size_t)b * NH + n) * HH * SKL;
    const bf16* qrow = Qb + (((size_t)b * NH + n) * SQL + q0 + l16) * HH;

    v16bf aQ0 = load_afrag(qrow, half);           // h 0..31
    v16bf aQ1 = load_afrag(qrow + 32, half);      // h 32..63

    v8f acc[4];
#pragma unroll
    for (int t = 0; t < 4; ++t) acc[t] = v8f_zero();
    float mrow[8], lrow[8];
#pragma unroll
    for (int r = 0; r < 8; ++r) { mrow[r] = -__builtin_inff(); lrow[r] = 0.f; }

    // Staging assignment: 16B per thread per tile (4KB tiles, 256 threads).
    const int krow_s = tid >> 3;                  // 0..31 (key within chunk)
    const int kcol_h = (tid & 7) * 8;             // 0,8,..,56
    const int vrow_h = tid >> 2;                  // 0..63 (h)
    const int vcol_s = (tid & 3) * 8;             // 0,8,16,24

    const int kblockmax = qb * 128 + 128;         // block-uniform chunk bound
    const int kmax = q0 + 16;                     // this wave's causal bound

    // Prologue: stage chunk 0 into buffer 0.
    async_ld16(&kbuf[0][krow_s][kcol_h], Khead + (size_t)krow_s * HH + kcol_h);
    async_ld16(&vbuf[0][vrow_h][vcol_s], Vhead + (size_t)vrow_h * SKL + vcol_s);
    asm volatile("s_wait_asynccnt 0x0" ::: "memory");
    __syncthreads();

    int p = 0;
    for (int k0 = 0; k0 < kblockmax; k0 += 32, p ^= 1) {
        // Kick off DMA for the next chunk while we compute this one.
        const int k1 = k0 + 32;
        if (k1 < kblockmax) {
            async_ld16(&kbuf[p ^ 1][krow_s][kcol_h],
                       Khead + (size_t)(k1 + krow_s) * HH + kcol_h);
            async_ld16(&vbuf[p ^ 1][vrow_h][vcol_s],
                       Vhead + (size_t)vrow_h * SKL + k1 + vcol_s);
        }

        if (k0 < kmax) {                          // wave-uniform predicate
            // ---- S = Q K^T for two 16-column tiles (contract over H=64) ----
            const bf16* kr0 = &kbuf[p][l16][0];
            const bf16* kr1 = &kbuf[p][16 + l16][0];
            v8f S0 = wmma_bf16(aQ0, ld_lds_frag(kr0 + half * 16), v8f_zero());
            S0 = wmma_bf16(aQ1, ld_lds_frag(kr0 + 32 + half * 16), S0);
            v8f S1 = wmma_bf16(aQ0, ld_lds_frag(kr1 + half * 16), v8f_zero());
            S1 = wmma_bf16(aQ1, ld_lds_frag(kr1 + 32 + half * 16), S1);

            // ---- causal mask + online softmax (fp32 half-wave butterflies) --
#pragma unroll
            for (int r = 0; r < 8; ++r) {
                const int row = q0 + r + 8 * half;
                if (k0 + l16 > row)      S0[r] = -__builtin_inff();
                if (k0 + 16 + l16 > row) S1[r] = -__builtin_inff();

                float mloc = fmaxf(S0[r], S1[r]);
#pragma unroll
                for (int m = 1; m < 16; m <<= 1)
                    mloc = fmaxf(mloc, __shfl_xor(mloc, m, 32));
                float mn = fmaxf(mrow[r], mloc);
                float sc = __expf(mrow[r] - mn);
                float e0 = __expf(S0[r] - mn);
                float e1 = __expf(S1[r] - mn);
                float ps = e0 + e1;
#pragma unroll
                for (int m = 1; m < 16; m <<= 1) ps += __shfl_xor(ps, m, 32);
                lrow[r] = lrow[r] * sc + ps;
                mrow[r] = mn;
                acc[0][r] *= sc; acc[1][r] *= sc;
                acc[2][r] *= sc; acc[3][r] *= sc;
                plds[wave][r + 8 * half][l16] = (bf16)e0;
                plds[wave][r + 8 * half][16 + l16] = (bf16)e1;
            }
            // LDS ops are in-order per wave; make the dependency explicit.
            asm volatile("s_wait_dscnt 0" ::: "memory");
            v16bf aP = cat8(*(const v8bf*)&plds[wave][l16][half * 16],
                            *(const v8bf*)&plds[wave][l16][half * 16 + 8]);

            // ---- acc += P * V (contract over 32 keys, 4 h-tiles) ----
#pragma unroll
            for (int ht = 0; ht < 4; ++ht)
                acc[ht] = wmma_bf16(
                    aP, ld_lds_frag(&vbuf[p][ht * 16 + l16][half * 16]),
                    acc[ht]);
        }

        // All of this wave's DMA done, then block-wide rendezvous so the
        // buffer we just filled is valid for everyone (and the one we will
        // overwrite next iteration is no longer being read).
        asm volatile("s_wait_asynccnt 0x0" ::: "memory");
        __syncthreads();
    }

    float rinv[8];
#pragma unroll
    for (int r = 0; r < 8; ++r) rinv[r] = 1.0f / lrow[r];
#pragma unroll
    for (int ht = 0; ht < 4; ++ht)
#pragma unroll
        for (int r = 0; r < 8; ++r) {
            size_t row = (size_t)b * SQL + q0 + r + 8 * half;
            zb[row * (NH * HH) + n * HH + ht * 16 + l16] =
                (bf16)(acc[ht][r] * rinv[r]);
        }
}

// ---------------------------------------------------------------------------
// Stage 3: out = z @ W_O + b_O   (M=4096, N=1024, K=1024)
// ---------------------------------------------------------------------------
__global__ __launch_bounds__(256) void k_out(const bf16* __restrict__ zb,
                                             const bf16* __restrict__ WOb,
                                             const float* __restrict__ bO,
                                             float* __restrict__ outp) {
    const int rowTile = blockIdx.x;
    const int lane = threadIdx.x & 31;
    const int wave = threadIdx.x >> 5;
    const int half = lane >> 4;
    const int l16 = lane & 15;

    const bf16* arow = zb + (size_t)(rowTile * 16 + l16) * (NH * HH);
    v8f acc[8];
#pragma unroll
    for (int t = 0; t < 8; ++t) acc[t] = v8f_zero();

    for (int k0 = 0; k0 < NH * HH; k0 += 32) {
        __builtin_prefetch(arow + k0 + 128, 0, 0);
        v16bf a = load_afrag(arow + k0, half);
#pragma unroll
        for (int t = 0; t < 8; ++t) {
            int c = wave * 128 + t * 16 + l16;
            v16bf bf = *(const v16bf*)(WOb + (size_t)c * (NH * HH) + k0 + half * 16);
            acc[t] = wmma_bf16(a, bf, acc[t]);
        }
    }
#pragma unroll
    for (int t = 0; t < 8; ++t) {
        int c = wave * 128 + t * 16 + l16;
        float bo = bO[c];
#pragma unroll
        for (int r = 0; r < 8; ++r) {
            int row = rowTile * 16 + r + 8 * half;
            outp[(size_t)row * DD + c] = acc[t][r] + bo;
        }
    }
}

// ---------------------------------------------------------------------------
extern "C" void kernel_launch(void* const* d_in, const int* in_sizes, int n_in,
                              void* d_out, int out_size, void* d_ws,
                              size_t ws_size, hipStream_t stream) {
    const float* x_q = (const float*)d_in[0];
    const float* x_kv = (const float*)d_in[1];
    // d_in[2] = bool mask: causal, applied analytically in k_attn.
    const float* W_Q = (const float*)d_in[3];
    const float* W_K = (const float*)d_in[4];
    const float* W_V = (const float*)d_in[5];
    const float* W_O = (const float*)d_in[6];
    const float* b_Q = (const float*)d_in[7];
    const float* b_K = (const float*)d_in[8];
    const float* b_V = (const float*)d_in[9];
    const float* b_O = (const float*)d_in[10];
    const float* ln1_g = (const float*)d_in[11];
    const float* ln1_b = (const float*)d_in[12];
    const float* ln2_g = (const float*)d_in[13];
    const float* ln2_b = (const float*)d_in[14];

    float* out_o = (float*)d_out;                    // [B,SQ,D]
    float* out_k = out_o + (size_t)BB * SQL * DD;    // [B,SK,N,H]
    float* out_v = out_k + (size_t)BB * SKL * NH * HH;

    // Workspace layout (bf16 buffers), ~56 MB total.
    char* ws = (char*)d_ws;
    const size_t ACT = (size_t)BB * SQL * DD;        // 4 Mi elements
    bf16* xqb = (bf16*)ws;
    bf16* xkvb = xqb + ACT;
    bf16* Wb = xkvb + ACT;                           // 3 x 1024 x 1024
    bf16* WOb = Wb + (size_t)3 * DD * NH * HH;       // 1024 x 1024
    bf16* Qb = WOb + (size_t)DD * NH * HH;           // [B,N,SQ,H]
    bf16* Kb = Qb + ACT;
    bf16* Vt = Kb + ACT;                             // [B,N,H,SK]
    bf16* zb = Vt + ACT;                             // [B*SQ, N*H]

    k_cvt_bf16<<<4096, 256, 0, stream>>>(x_q, xqb, (int)ACT);
    k_cvt_bf16<<<4096, 256, 0, stream>>>(x_kv, xkvb, (int)ACT);
    k_pack_w<<<dim3(4096, 3), 256, 0, stream>>>(W_Q, W_K, W_V, Wb);
    k_pack_wo<<<4096, 256, 0, stream>>>(W_O, WOb);

    k_proj<<<dim3(256, 3), 256, 0, stream>>>(xqb, xkvb, Wb, b_Q, b_K, b_V,
                                             ln1_g, ln1_b, ln2_g, ln2_b,
                                             out_k, out_v, Qb, Kb, Vt);
    k_attn<<<512, 256, 0, stream>>>(Qb, Kb, Vt, zb);
    k_out<<<256, 256, 0, stream>>>(zb, WOb, b_O, out_o);
}
